// CausalMultiHeadSelfAttention_4389456577095
// MI455X (gfx1250) — matrix operand order repaired and checked by reading, not compiled, not run on hardware
//
#include <hip/hip_runtime.h>
#include <math.h>

typedef _Float16 v8h  __attribute__((ext_vector_type(8)));
typedef _Float16 v16h __attribute__((ext_vector_type(16)));
typedef float    v8f  __attribute__((ext_vector_type(8)));

#define D_MODEL  1024
#define N_HEADS  16
#define HEAD_DIM 64
#define SEQ      2048
#define BATCH    4
#define MROWS    (BATCH * SEQ)
#define AW       4   // attention waves per block

__device__ __forceinline__ v16h cat8(v8h lo, v8h hi) {
  return __builtin_shufflevector(lo, hi, 0,1,2,3,4,5,6,7,8,9,10,11,12,13,14,15);
}

// ---------------- fp32 -> f16 conversion ----------------
__global__ __launch_bounds__(256) void cvt_f32_f16(const float* __restrict__ src,
                                                   _Float16* __restrict__ dst, int n) {
  int i = blockIdx.x * blockDim.x + threadIdx.x;
  int stride = gridDim.x * blockDim.x;
  for (; i < n; i += stride) dst[i] = (_Float16)src[i];
}

// ---------------- WMMA GEMM: C[M,1024] = A[M,1024] * W[1024,1024]^T ----------------
// 64x64 output per wave = 4x4 grid of 16x16 WMMA subtiles (16 wmma / 8 frag loads per k-step).
// mode 0: fp32 row-major -> outF
// mode 1: RoPE + scale -> f16 outH laid out [B,H,S,64]
// mode 2: f16 outH laid out [B,H,64,S]   (transposed V)
__global__ __launch_bounds__(256) void gemm_qkv(const _Float16* __restrict__ A,
                                                const _Float16* __restrict__ W,
                                                float* __restrict__ outF,
                                                _Float16* __restrict__ outH,
                                                int mode, float scale) {
  const int N = D_MODEL;
  int lane = threadIdx.x & 31;
  int wave = threadIdx.x >> 5;
  int tile = blockIdx.x * 8 + wave;       // one 64x64 output macro-tile per wave
  const int tilesN = N >> 6;              // 16
  int mT = tile / tilesN;
  int nT = tile - mT * tilesN;
  int half = lane >> 4, nl = lane & 15;
  int mBase = mT * 64;
  int nBase = nT * 64;

  const _Float16* Ar[4];
  const _Float16* Wr[4];
#pragma unroll
  for (int i = 0; i < 4; ++i) {
    Ar[i] = A + (size_t)(mBase + 16 * i + nl) * D_MODEL;
    Wr[i] = W + (size_t)(nBase + 16 * i + nl) * D_MODEL;
  }

  v8f acc[4][4];
#pragma unroll
  for (int i = 0; i < 4; ++i)
#pragma unroll
    for (int j = 0; j < 4; ++j) acc[i][j] = (v8f){};

  for (int k0 = 0; k0 < D_MODEL; k0 += 32) {
    v16h af[4], bf[4];
#pragma unroll
    for (int i = 0; i < 4; ++i) {
      // A fragment (16x32 f16): lane row = nl, K = 16*half + i
      v8h alo = *(const v8h*)(Ar[i] + k0 + 16 * half);
      v8h ahi = *(const v8h*)(Ar[i] + k0 + 16 * half + 8);
      af[i] = cat8(alo, ahi);
    }
#pragma unroll
    for (int j = 0; j < 4; ++j) {
      // B fragment (32x16 f16): B[k][n] = W[n][k]; lane col = nl, K = 16*half + i
      bf[j] = *(const v16h*)(Wr[j] + k0 + 16 * half);
    }
#pragma unroll
    for (int i = 0; i < 4; ++i)
#pragma unroll
      for (int j = 0; j < 4; ++j)
        acc[i][j] = __builtin_amdgcn_wmma_f32_16x16x32_f16(false, af[i], false, bf[j],
                                                           (short)0, acc[i][j], false, false);
  }

  int b = mBase / SEQ;                    // 64 rows never straddle a batch (2048 % 64 == 0)
  int sBase = mBase - b * SEQ;

  if (mode == 0) {
#pragma unroll
    for (int i = 0; i < 4; ++i)
#pragma unroll
      for (int j = 0; j < 4; ++j) {
        int nG = nBase + 16 * j + nl;
#pragma unroll
        for (int r = 0; r < 8; ++r) {
          int rg = mBase + 16 * i + r + 8 * half;
          outF[(size_t)rg * N + nG] = acc[i][j][r];
        }
      }
  } else if (mode == 1) {
#pragma unroll
    for (int j = 0; j < 4; ++j) {
      int nG = nBase + 16 * j + nl;
      int h = nG >> 6, dd = nG & 63;
      float invf = __powf(10000.0f, -(float)(dd & ~1) / 64.0f);
#pragma unroll
      for (int i = 0; i < 4; ++i) {
#pragma unroll
        for (int r = 0; r < 8; ++r) {
          int s = sBase + 16 * i + r + 8 * half;
          float v = acc[i][j][r];
          float p = __shfl_xor(v, 1);     // RoPE partner (even<->odd feature)
          float sn, cs;
          __sincosf((float)s * invf, &sn, &cs);
          float o = (dd & 1) ? (v * cs + p * sn) : (v * cs - p * sn);
          o *= scale;
          outH[(((size_t)(b * N_HEADS + h) * SEQ) + s) * HEAD_DIM + dd] = (_Float16)o;
        }
      }
    }
  } else {
#pragma unroll
    for (int j = 0; j < 4; ++j) {
      int nG = nBase + 16 * j + nl;
      int h = nG >> 6, dd = nG & 63;
#pragma unroll
      for (int i = 0; i < 4; ++i) {
#pragma unroll
        for (int r = 0; r < 8; ++r) {
          int s = sBase + 16 * i + r + 8 * half;
          outH[(((size_t)(b * N_HEADS + h) * HEAD_DIM) + dd) * SEQ + s] = (_Float16)acc[i][j][r];
        }
      }
    }
  }
}

// ---------------- Flash attention: one wave per 16-query tile ----------------
__global__ __launch_bounds__(32 * AW) void attn_fwd(const _Float16* __restrict__ qb,
                                                    const _Float16* __restrict__ kb,
                                                    const _Float16* __restrict__ vt,
                                                    _Float16* __restrict__ ctxb) {
  __shared__ _Float16 plds[AW][16 * 32];
  int lane = threadIdx.x & 31;
  int wave = threadIdx.x >> 5;
  int bh = blockIdx.y;
  int qt = blockIdx.x * AW + wave;
  int qbase = qt * 16;
  int half = lane >> 4, nl = lane & 15;

  const _Float16* Q  = qb + (size_t)bh * SEQ * HEAD_DIM;
  const _Float16* Kp = kb + (size_t)bh * SEQ * HEAD_DIM;
  const _Float16* Vp = vt + (size_t)bh * HEAD_DIM * SEQ;

  // Q fragments: A layout over dk 0..31 and 32..63 (scale already folded in)
  const _Float16* Qrow = Q + (size_t)(qbase + nl) * HEAD_DIM;
  v16h qf0 = cat8(*(const v8h*)(Qrow + 16 * half), *(const v8h*)(Qrow + 16 * half + 8));
  v16h qf1 = cat8(*(const v8h*)(Qrow + 32 + 16 * half), *(const v8h*)(Qrow + 40 + 16 * half));

  v8f c0 = {}, c1 = {}, c2 = {}, c3 = {};   // ctx accumulator 16x64
  float mrow[8], lrow[8];
#pragma unroll
  for (int r = 0; r < 8; ++r) { mrow[r] = -1e30f; lrow[r] = 0.0f; }

  int jmax = (qbase + 15) >> 5;             // causal: key tiles of 32
  for (int j = 0; j <= jmax; ++j) {
    int kb0 = j * 32;
    // logits tiles: S0 = keys kb0..kb0+15, S1 = keys kb0+16..kb0+31
    v8f s0 = {}, s1 = {};
    {
      const _Float16* Kr = Kp + (size_t)(kb0 + nl) * HEAD_DIM;
      v16h kfa = *(const v16h*)(Kr + 16 * half);
      v16h kfb = *(const v16h*)(Kr + 32 + 16 * half);
      s0 = __builtin_amdgcn_wmma_f32_16x16x32_f16(false, qf0, false, kfa, (short)0, s0, false, false);
      s0 = __builtin_amdgcn_wmma_f32_16x16x32_f16(false, qf1, false, kfb, (short)0, s0, false, false);
    }
    {
      const _Float16* Kr = Kp + (size_t)(kb0 + 16 + nl) * HEAD_DIM;
      v16h kfa = *(const v16h*)(Kr + 16 * half);
      v16h kfb = *(const v16h*)(Kr + 32 + 16 * half);
      s1 = __builtin_amdgcn_wmma_f32_16x16x32_f16(false, qf0, false, kfa, (short)0, s1, false, false);
      s1 = __builtin_amdgcn_wmma_f32_16x16x32_f16(false, qf1, false, kfb, (short)0, s1, false, false);
    }
    // causal mask (only the trailing tiles need it)
    if (kb0 + 31 > qbase) {
#pragma unroll
      for (int r = 0; r < 8; ++r) {
        int qg = qbase + r + 8 * half;
        if (kb0 + nl > qg)      s0[r] = -1e30f;
        if (kb0 + 16 + nl > qg) s1[r] = -1e30f;
      }
    }
    // online softmax: row max / row sum across the 16 lanes of each half
    float rmax[8];
#pragma unroll
    for (int r = 0; r < 8; ++r) rmax[r] = fmaxf(s0[r], s1[r]);
#pragma unroll
    for (int r = 0; r < 8; ++r)
      for (int off = 8; off; off >>= 1)
        rmax[r] = fmaxf(rmax[r], __shfl_xor(rmax[r], off));
#pragma unroll
    for (int r = 0; r < 8; ++r) {
      float mnew  = fmaxf(mrow[r], rmax[r]);
      float alpha = __expf(mrow[r] - mnew);
      mrow[r] = mnew;
      float p0 = __expf(s0[r] - mnew);
      float p1 = __expf(s1[r] - mnew);
      s0[r] = p0; s1[r] = p1;
      float ps = p0 + p1;
      for (int off = 8; off; off >>= 1) ps += __shfl_xor(ps, off);
      lrow[r] = lrow[r] * alpha + ps;
      c0[r] *= alpha; c1[r] *= alpha; c2[r] *= alpha; c3[r] *= alpha;
    }
    // stage P (16x32 f16) through LDS: D layout -> A layout
#pragma unroll
    for (int r = 0; r < 8; ++r) {
      int m = r + 8 * half;
      plds[wave][m * 32 + nl]      = (_Float16)s0[r];
      plds[wave][m * 32 + 16 + nl] = (_Float16)s1[r];
    }
    asm volatile("s_wait_dscnt 0" ::: "memory");   // cross-lane LDS RAW within wave
    v16h pf = cat8(*(const v8h*)&plds[wave][nl * 32 + 16 * half],
                   *(const v8h*)&plds[wave][nl * 32 + 16 * half + 8]);
    // PV: V^T layout [64][S] -> B fragment is 16 contiguous f16 per lane
    c0 = __builtin_amdgcn_wmma_f32_16x16x32_f16(false, pf, false,
           *(const v16h*)(Vp + (size_t)(0 * 16 + nl) * SEQ + kb0 + 16 * half), (short)0, c0, false, false);
    c1 = __builtin_amdgcn_wmma_f32_16x16x32_f16(false, pf, false,
           *(const v16h*)(Vp + (size_t)(1 * 16 + nl) * SEQ + kb0 + 16 * half), (short)0, c1, false, false);
    c2 = __builtin_amdgcn_wmma_f32_16x16x32_f16(false, pf, false,
           *(const v16h*)(Vp + (size_t)(2 * 16 + nl) * SEQ + kb0 + 16 * half), (short)0, c2, false, false);
    c3 = __builtin_amdgcn_wmma_f32_16x16x32_f16(false, pf, false,
           *(const v16h*)(Vp + (size_t)(3 * 16 + nl) * SEQ + kb0 + 16 * half), (short)0, c3, false, false);
  }

  // normalize and write ctx as f16 [B,S,D]
  int b = bh / N_HEADS, h = bh % N_HEADS;
#pragma unroll
  for (int r = 0; r < 8; ++r) {
    float inv = 1.0f / lrow[r];
    int s = qbase + r + 8 * half;
    size_t base = ((size_t)b * SEQ + s) * D_MODEL + h * HEAD_DIM;
    ctxb[base + 0 * 16 + nl] = (_Float16)(c0[r] * inv);
    ctxb[base + 1 * 16 + nl] = (_Float16)(c1[r] * inv);
    ctxb[base + 2 * 16 + nl] = (_Float16)(c2[r] * inv);
    ctxb[base + 3 * 16 + nl] = (_Float16)(c3[r] * inv);
  }
}

// ---------------- host launcher ----------------
extern "C" void kernel_launch(void* const* d_in, const int* in_sizes, int n_in,
                              void* d_out, int out_size, void* d_ws, size_t ws_size,
                              hipStream_t stream) {
  const float* x  = (const float*)d_in[0];
  const float* wq = (const float*)d_in[1];
  const float* wk = (const float*)d_in[2];
  const float* wv = (const float*)d_in[3];
  const float* wo = (const float*)d_in[4];
  float* out = (float*)d_out;

  char* ws = (char*)d_ws;
  size_t off = 0;
  auto alloc = [&](size_t bytes) -> void* {
    void* p = ws + off;
    off += (bytes + 255) & ~(size_t)255;
    return p;
  };
  const size_t MD = (size_t)MROWS * D_MODEL;    // 8.4M elems
  const size_t WD = (size_t)D_MODEL * D_MODEL;  // 1.05M elems

  _Float16* xh   = (_Float16*)alloc(MD * 2);
  _Float16* wqh  = (_Float16*)alloc(WD * 2);
  _Float16* wkh  = (_Float16*)alloc(WD * 2);
  _Float16* wvh  = (_Float16*)alloc(WD * 2);
  _Float16* woh  = (_Float16*)alloc(WD * 2);
  _Float16* qh   = (_Float16*)alloc(MD * 2);    // [B,H,S,64]
  _Float16* kh   = (_Float16*)alloc(MD * 2);    // [B,H,S,64]
  _Float16* vth  = (_Float16*)alloc(MD * 2);    // [B,H,64,S]
  _Float16* ctxh = (_Float16*)alloc(MD * 2);    // [B,S,D]

  cvt_f32_f16<<<2048, 256, 0, stream>>>(x,  xh,  (int)MD);
  cvt_f32_f16<<<512,  256, 0, stream>>>(wq, wqh, (int)WD);
  cvt_f32_f16<<<512,  256, 0, stream>>>(wk, wkh, (int)WD);
  cvt_f32_f16<<<512,  256, 0, stream>>>(wv, wvh, (int)WD);
  cvt_f32_f16<<<512,  256, 0, stream>>>(wo, woh, (int)WD);

  int gemmGrid = (MROWS / 64) * (D_MODEL / 64) / 8;  // 256 blocks, 8 waves each
  gemm_qkv<<<gemmGrid, 256, 0, stream>>>(xh, wqh, nullptr, qh,  1, 0.125f); // Q: RoPE + 1/sqrt(dk)
  gemm_qkv<<<gemmGrid, 256, 0, stream>>>(xh, wkh, nullptr, kh,  1, 1.0f);   // K: RoPE
  gemm_qkv<<<gemmGrid, 256, 0, stream>>>(xh, wvh, nullptr, vth, 2, 1.0f);   // V: transpose

  dim3 ag(SEQ / 16 / AW, BATCH * N_HEADS);
  attn_fwd<<<ag, 32 * AW, 0, stream>>>(qh, kh, vth, ctxh);

  gemm_qkv<<<gemmGrid, 256, 0, stream>>>(ctxh, woh, out, nullptr, 0, 1.0f); // O projection
}
